// InfoMultiAttn_79559974191127
// MI455X (gfx1250) — matrix operand order repaired and checked by reading, not compiled, not run on hardware
//
#include <hip/hip_runtime.h>
#include <hip/hip_bf16.h>

typedef __attribute__((ext_vector_type(16))) __bf16 v16bf;
typedef __attribute__((ext_vector_type(8)))  float  v8f;

#define B_   8192
#define L_   64
#define E_   128
#define H_   8
#define HD_  16
#define XS   136   // padded bf16 row stride for LDS tiles

union BfVec { v16bf v; uint4 q[2]; };
union F8    { v8f   v; float f[8]; };

__device__ __forceinline__ unsigned short f2bf(float f) {
    unsigned int u = __float_as_uint(f);
    u += 0x7FFFu + ((u >> 16) & 1u);       // round-to-nearest-even
    return (unsigned short)(u >> 16);
}
__device__ __forceinline__ unsigned int pack2bf(float lo, float hi) {
    return (unsigned int)f2bf(lo) | ((unsigned int)f2bf(hi) << 16);
}
__device__ __forceinline__ float bf2f(unsigned short h) {
    return __uint_as_float(((unsigned int)h) << 16);
}
__device__ __forceinline__ float bfu_lo(unsigned int u) { return __uint_as_float(u << 16); }
__device__ __forceinline__ float bfu_hi(unsigned int u) { return __uint_as_float(u & 0xffff0000u); }

// ---- Kernel 1: convert Wk,Wv (rows E..3E of in_proj_w) to bf16 in workspace ----
__global__ void convert_wkv_kernel(const float* __restrict__ in_proj_w,
                                   unsigned short* __restrict__ wkv) {
    int i = blockIdx.x * 256 + threadIdx.x;
    if (i < 2 * E_ * E_) wkv[i] = f2bf(in_proj_w[E_ * E_ + i]);
}

// ---- Kernel 2: one block (8 wave32) per batch element ----
__global__ void __launch_bounds__(256)
info_attn_kernel(const float* __restrict__ inter_info,
                 const int*   __restrict__ lengths,
                 const float* __restrict__ in_proj_w,
                 const float* __restrict__ in_proj_b,
                 const float* __restrict__ out_proj_w,
                 const float* __restrict__ out_proj_b,
                 const unsigned short* __restrict__ wkv,   // [Wk | Wv] bf16
                 float* __restrict__ out) {
    __shared__ unsigned short Xbf[L_ * XS];   // input tile, bf16
    __shared__ unsigned short Kf16[L_ * XS];  // K = X @ Wk^T + bk, bf16
    __shared__ unsigned short Vf16[L_ * XS];  // V = X @ Wv^T + bv, bf16
    __shared__ float X0f[E_];                 // query row, f32
    __shared__ float qs[E_];                  // q vector
    __shared__ float attnL[H_ * L_];          // softmax weights
    __shared__ float ctxL[E_];                // attention context

    const int b    = blockIdx.x;
    const int t    = threadIdx.x;
    const int w    = t >> 5;                  // wave id 0..7
    const int lane = t & 31;
    const float* Xg = inter_info + (size_t)b * L_ * E_;

    // ---- Phase A: stage X into LDS as bf16 (vectorized), query row as f32 ----
    {
        const float4* Xg4 = (const float4*)Xg;
        for (int i = t; i < (L_ * E_) / 4; i += 256) {     // 8 iters
            int r = i >> 5, c4 = i & 31;                   // row, float4-col
            float4 x = Xg4[i];
            unsigned int* dst = (unsigned int*)&Xbf[r * XS + c4 * 4];
            dst[0] = pack2bf(x.x, x.y);
            dst[1] = pack2bf(x.z, x.w);
        }
        if (t < E_) X0f[t] = Xg[t];
    }
    __syncthreads();

    // ---- Phase B: K and V projections via WMMA (bf16 -> f32 acc) ----
    const int lt    = w >> 1;                 // l-tile 0..3
    const int ehalf = w & 1;                  // which 64 output features
    const int mrow  = lt * 16 + (lane & 15);
    const int koffA = (lane & 16) ? 16 : 0;
    const int koffB = (lane & 16) ? 16 : 0;
    const int ecol  = lane & 15;

    F8 accK[4], accV[4];
    for (int j = 0; j < 4; ++j) {
        int e = ehalf * 64 + j * 16 + ecol;
        float bkv = in_proj_b[E_ + e];
        float bvv = in_proj_b[2 * E_ + e];
        for (int r = 0; r < 8; ++r) { accK[j].f[r] = bkv; accV[j].f[r] = bvv; }
    }
    #pragma unroll
    for (int cc = 0; cc < 4; ++cc) {
        const int c0 = cc * 32;
        // A fragment (shared by K and V matmuls)
        BfVec a;
        const unsigned short* xr = &Xbf[mrow * XS + c0 + koffA];
        a.q[0] = *(const uint4*)xr;
        a.q[1] = *(const uint4*)(xr + 8);
        // Preload ALL 8 B fragments of this chunk, then issue 8 WMMAs back-to-back
        BfVec bfr[8];
        #pragma unroll
        for (int j = 0; j < 4; ++j) {
            int e = ehalf * 64 + j * 16 + ecol;
            const unsigned short* wkr = &wkv[e * E_ + c0 + koffB];
            bfr[j].q[0]     = *(const uint4*)wkr;
            bfr[j].q[1]     = *(const uint4*)(wkr + 8);
            const unsigned short* wvr = wkr + E_ * E_;
            bfr[4 + j].q[0] = *(const uint4*)wvr;
            bfr[4 + j].q[1] = *(const uint4*)(wvr + 8);
        }
        #pragma unroll
        for (int j = 0; j < 4; ++j)
            accK[j].v = __builtin_amdgcn_wmma_f32_16x16x32_bf16(
                false, a.v, false, bfr[j].v, (short)0, accK[j].v, false, false);
        #pragma unroll
        for (int j = 0; j < 4; ++j)
            accV[j].v = __builtin_amdgcn_wmma_f32_16x16x32_bf16(
                false, a.v, false, bfr[4 + j].v, (short)0, accV[j].v, false, false);
    }
    // store D tiles to LDS per 16x16 f32 C/D layout
    {
        const int rbase = lt * 16 + ((lane & 16) ? 8 : 0);
        for (int j = 0; j < 4; ++j) {
            int ncol = ehalf * 64 + j * 16 + ecol;
            for (int r = 0; r < 8; ++r) {
                Kf16[(rbase + r) * XS + ncol] = f2bf(accK[j].f[r]);
                Vf16[(rbase + r) * XS + ncol] = f2bf(accV[j].f[r]);
            }
        }
    }
    __syncthreads();

    // ---- Phase C: q = Wq @ x0 + bq (threads 0..127); query_info copy (128..255) ----
    if (t < E_) {
        const float4* wq4 = (const float4*)(in_proj_w + (size_t)t * E_);
        float acc = in_proj_b[t];
        #pragma unroll 8
        for (int c4 = 0; c4 < E_ / 4; ++c4) {
            float4 wv = wq4[c4];
            acc += X0f[c4 * 4 + 0] * wv.x + X0f[c4 * 4 + 1] * wv.y
                 + X0f[c4 * 4 + 2] * wv.z + X0f[c4 * 4 + 3] * wv.w;
        }
        qs[t] = acc;
    } else {
        int e = t - E_;
        out[(size_t)B_ * E_ + (size_t)b * E_ + e] = Xg[e];  // query_info output
    }
    __syncthreads();

    // ---- Phase D: scores + masked softmax (wave w == head h) ----
    {
        const int h   = w;
        const int len = lengths[b];
        float s0 = 0.f, s1 = 0.f;
        const unsigned int* k0 = (const unsigned int*)&Kf16[(lane)      * XS + h * HD_];
        const unsigned int* k1 = (const unsigned int*)&Kf16[(lane + 32) * XS + h * HD_];
        unsigned int kr0[8], kr1[8];
        *(uint4*)(kr0)     = ((const uint4*)k0)[0];
        *(uint4*)(kr0 + 4) = ((const uint4*)k0)[1];
        *(uint4*)(kr1)     = ((const uint4*)k1)[0];
        *(uint4*)(kr1 + 4) = ((const uint4*)k1)[1];
        #pragma unroll
        for (int i = 0; i < 8; ++i) {
            float qa = qs[h * HD_ + 2 * i];
            float qb = qs[h * HD_ + 2 * i + 1];
            s0 += qa * bfu_lo(kr0[i]) + qb * bfu_hi(kr0[i]);
            s1 += qa * bfu_lo(kr1[i]) + qb * bfu_hi(kr1[i]);
        }
        s0 *= 0.25f;  s1 *= 0.25f;            // 1/sqrt(HD)
        if (lane      >= len) s0 = -3.0e38f;
        if (lane + 32 >= len) s1 = -3.0e38f;
        float m = fmaxf(s0, s1);
        for (int off = 16; off > 0; off >>= 1) m = fmaxf(m, __shfl_xor(m, off));
        float e0 = __expf(s0 - m), e1 = __expf(s1 - m);
        float sum = e0 + e1;
        for (int off = 16; off > 0; off >>= 1) sum += __shfl_xor(sum, off);
        float inv = 1.0f / sum;
        attnL[h * L_ + lane]      = e0 * inv;
        attnL[h * L_ + lane + 32] = e1 * inv;
    }
    __syncthreads();

    // ---- Phase E: ctx[h][d] = sum_l attn[h][l] * V[l][h*16+d] ----
    if (t < E_) {
        const int h = t >> 4, d = t & 15;
        float acc = 0.f;
        for (int l = 0; l < L_; ++l)
            acc += attnL[h * L_ + l] * bf2f(Vf16[l * XS + h * HD_ + d]);
        ctxL[t] = acc;
    }
    __syncthreads();

    // ---- Phase F: out = Wo @ ctx + bo ----
    if (t < E_) {
        const float4* wo4 = (const float4*)(out_proj_w + (size_t)t * E_);
        float acc = out_proj_b[t];
        #pragma unroll 8
        for (int c4 = 0; c4 < E_ / 4; ++c4) {
            float4 wv = wo4[c4];
            acc += ctxL[c4 * 4 + 0] * wv.x + ctxL[c4 * 4 + 1] * wv.y
                 + ctxL[c4 * 4 + 2] * wv.z + ctxL[c4 * 4 + 3] * wv.w;
        }
        out[(size_t)b * E_ + t] = acc;
    }
}

extern "C" void kernel_launch(void* const* d_in, const int* in_sizes, int n_in,
                              void* d_out, int out_size, void* d_ws, size_t ws_size,
                              hipStream_t stream) {
    (void)in_sizes; (void)n_in; (void)out_size; (void)ws_size;
    const float* inter_info = (const float*)d_in[0];
    const int*   lengths    = (const int*)d_in[1];
    const float* in_proj_w  = (const float*)d_in[2];
    const float* in_proj_b  = (const float*)d_in[3];
    const float* out_proj_w = (const float*)d_in[4];
    const float* out_proj_b = (const float*)d_in[5];
    float* out = (float*)d_out;
    unsigned short* wkv = (unsigned short*)d_ws;   // 64 KB: bf16 Wk|Wv

    convert_wkv_kernel<<<(2 * E_ * E_ + 255) / 256, 256, 0, stream>>>(in_proj_w, wkv);
    info_attn_kernel<<<B_, 256, 0, stream>>>(inter_info, lengths, in_proj_w, in_proj_b,
                                             out_proj_w, out_proj_b, wkv, out);
}
